// MixtralDecoderLayer_37709812859145
// MI455X (gfx1250) — matrix operand order repaired and checked by reading, not compiled, not run on hardware
//
#include <hip/hip_runtime.h>
#include <hip/hip_bf16.h>

#define TT 2048
#define HH 2048
#define NHEADS 16
#define NKVH 4
#define HDIM 128
#define NEXP 8
#define DFF 4096

typedef __attribute__((ext_vector_type(16))) __bf16 v16bf;
typedef __attribute__((ext_vector_type(8))) float v8f;
typedef __attribute__((ext_vector_type(8))) unsigned int v8u;
typedef __attribute__((ext_vector_type(4))) unsigned int v4u;
typedef __attribute__((ext_vector_type(2))) unsigned int v2u;

// fp32 -> bf16, round-half-up (matches RNE except exact ties; 1 VALU op)
__device__ __forceinline__ unsigned short f2bf(float f) {
  unsigned int u = __builtin_bit_cast(unsigned int, f);
  return (unsigned short)((u + 0x8000u) >> 16);
}

#if __has_builtin(__builtin_amdgcn_perm)
// Pack two bf16: two v_add + one v_perm_b32 selecting the high halves.
__device__ __forceinline__ unsigned int pack2bf(float a, float b) {
  unsigned int ua = __builtin_bit_cast(unsigned int, a) + 0x8000u;
  unsigned int ub = __builtin_bit_cast(unsigned int, b) + 0x8000u;
  // dst = { ub[31:16], ua[31:16] } : src0=ub (bytes 7:4), src1=ua (bytes 3:0)
  return __builtin_amdgcn_perm(ub, ua, 0x07060302u);
}
#else
__device__ __forceinline__ unsigned int pack2bf(float a, float b) {
  return (unsigned int)f2bf(a) | ((unsigned int)f2bf(b) << 16);
}
#endif

// A-fragment (16 x 32 bf16), interleaved layout per CDNA5 ISA 16-bit A table:
// lane m = lane&15; halves 0..7 -> K = koff + 8*hs + i ; halves 8..15 -> K = koff + 16 + 8*hs + (i-8)
// pitch and koff must be multiples of 8 halves -> two ds_load_b128 per fragment.
__device__ __forceinline__ v16bf lds_frag_a(const unsigned short* base, int row0, int pitch, int koff) {
  int lane = threadIdx.x & 31;
  int hs = lane >> 4, m = lane & 15;
  const unsigned short* p = base + (row0 + m) * pitch + koff + 8 * hs;
  v4u lo = *(const v4u*)(p);        // halves 0..7
  v4u hi = *(const v4u*)(p + 16);   // halves 8..15 (K + 16)
  v8u f = __builtin_shufflevector(lo, hi, 0, 1, 2, 3, 4, 5, 6, 7);
  return __builtin_bit_cast(v16bf, f);
}

// B-fragment (32 x 16 bf16): lane n = lane&15; halves 0..7 -> K = koff + 8*hs + i ; halves 8..15 -> K = koff + 16 + 8*hs + (i-8)
__device__ __forceinline__ v16bf lds_frag_b(const unsigned short* base, int row0, int pitch, int koff) {
  int lane = threadIdx.x & 31;
  int hs = lane >> 4, n = lane & 15;
  const unsigned short* p = base + (row0 + n) * pitch + koff + 8 * hs;
  v4u lo = *(const v4u*)(p);        // halves 0..7
  v4u hi = *(const v4u*)(p + 16);   // halves 8..15 (K + 16)
  v8u f = __builtin_shufflevector(lo, hi, 0, 1, 2, 3, 4, 5, 6, 7);
  return __builtin_bit_cast(v16bf, f);
}

#define WMMA_BF16(A, B, C) \
  __builtin_amdgcn_wmma_f32_16x16x32_bf16(false, (A), false, (B), (short)0, (C), false, false)

// ---------------------------------------------------------------------------
// GEMM: C[M,N] = A[M,K] @ B[K,N] (fp32 in, bf16 WMMA, f32 accumulate)
// Double-buffered LDS: next tile's global loads issue before this tile's WMMAs.
// mode 0: C = acc ; mode 1: C = acc + D ; mode 2: C += rowscale[row*rs]*acc
// M,N multiples of 128; K multiple of 32.
// ---------------------------------------------------------------------------
#define BM 128
#define BN 128
#define BK 32
#define GP 40  // LDS pitch in halves (32 + 8 pad; 80 B rows keep 16B alignment)

__global__ __launch_bounds__(256) void gemm_kernel(
    const float* __restrict__ A, const float* __restrict__ B, float* __restrict__ C,
    int M, int N, int K, int mode, const float* __restrict__ D,
    const float* __restrict__ rowscale, int rs_stride) {
  __shared__ alignas(16) unsigned short As[2][BM * GP];
  __shared__ alignas(16) unsigned short Bs[2][BN * GP];
  const int tid = threadIdx.x;
  const int lane = tid & 31;
  const int wave = tid >> 5;
  const int wr = wave >> 2, wc = wave & 3;  // 2x4 wave grid; wave tile 64x32
  const int bm = blockIdx.y * BM, bn = blockIdx.x * BN;

  // Fetch one k-slab (A:128x32, B:32x128) into registers (no wait yet).
  auto load_regs = [&](int k0, float4* ra, float4* rb) {
#pragma unroll
    for (int it = 0; it < 4; ++it) {
      int slot = tid + it * 256;           // 128 rows x 8 float4
      int row = slot >> 3, c4 = slot & 7;
      ra[it] = *(const float4*)(A + (size_t)(bm + row) * K + k0 + c4 * 4);
    }
#pragma unroll
    for (int it = 0; it < 4; ++it) {
      int slot = tid + it * 256;           // 32 k-rows x 32 float4
      int kk = slot >> 5, c4 = slot & 31;
      rb[it] = *(const float4*)(B + (size_t)(k0 + kk) * N + bn + c4 * 4);
    }
  };
  // Convert fp32->bf16 and store to LDS buffer `buf` (B transposed -> Bs[n][k]).
  auto store_lds = [&](int buf, const float4* ra, const float4* rb) {
#pragma unroll
    for (int it = 0; it < 4; ++it) {
      int slot = tid + it * 256;
      int row = slot >> 3, c4 = slot & 7;
      v2u pk = {pack2bf(ra[it].x, ra[it].y), pack2bf(ra[it].z, ra[it].w)};
      *(v2u*)&As[buf][row * GP + c4 * 4] = pk;  // 8B-aligned single store
    }
#pragma unroll
    for (int it = 0; it < 4; ++it) {
      int slot = tid + it * 256;
      int kk = slot >> 5, c4 = slot & 31;
      Bs[buf][(c4 * 4 + 0) * GP + kk] = f2bf(rb[it].x);
      Bs[buf][(c4 * 4 + 1) * GP + kk] = f2bf(rb[it].y);
      Bs[buf][(c4 * 4 + 2) * GP + kk] = f2bf(rb[it].z);
      Bs[buf][(c4 * 4 + 3) * GP + kk] = f2bf(rb[it].w);
    }
  };

  v8f acc[4][2];
#pragma unroll
  for (int mt = 0; mt < 4; ++mt)
#pragma unroll
    for (int nt = 0; nt < 2; ++nt)
#pragma unroll
      for (int i = 0; i < 8; ++i) acc[mt][nt][i] = 0.f;

  float4 ra[4], rb[4];
  load_regs(0, ra, rb);
  store_lds(0, ra, rb);
  __syncthreads();

  int cur = 0;
  for (int k0 = 0; k0 < K; k0 += BK) {
    const bool more = (k0 + BK) < K;
    if (more) load_regs(k0 + BK, ra, rb);  // in flight during WMMAs below

    v16bf af[4];
#pragma unroll
    for (int mt = 0; mt < 4; ++mt) af[mt] = lds_frag_a(&As[cur][0], wr * 64 + mt * 16, GP, 0);
#pragma unroll
    for (int nt = 0; nt < 2; ++nt) {
      v16bf bfm = lds_frag_b(&Bs[cur][0], wc * 32 + nt * 16, GP, 0);
#pragma unroll
      for (int mt = 0; mt < 4; ++mt)
        acc[mt][nt] = WMMA_BF16(af[mt], bfm, acc[mt][nt]);
    }

    if (more) store_lds(cur ^ 1, ra, rb);  // loadcnt wait lands here, after WMMAs
    __syncthreads();
    cur ^= 1;
  }

  const int col0 = lane & 15, hs = lane >> 4;
#pragma unroll
  for (int mt = 0; mt < 4; ++mt)
#pragma unroll
    for (int nt = 0; nt < 2; ++nt) {
      int col = bn + wc * 32 + nt * 16 + col0;
#pragma unroll
      for (int i = 0; i < 8; ++i) {
        int row = bm + wr * 64 + mt * 16 + i + 8 * hs;
        size_t idx = (size_t)row * N + col;
        float v = acc[mt][nt][i];
        if (mode == 0)      C[idx] = v;
        else if (mode == 1) C[idx] = v + D[idx];
        else                C[idx] += rowscale[(size_t)row * rs_stride] * v;
      }
    }
}

// ---------------------------------------------------------------------------
// Flash attention (causal, GQA rep=4). Block: 128 q-rows x 1 head, 8 waves,
// each wave owns 16 q-rows. Key tiles of 64, K/V staged to LDS as bf16.
// ---------------------------------------------------------------------------
#define AQ 128
#define AKT 64
#define KP 136  // Kt pitch (halves): [key][hd]; 272 B rows, 16B-aligned
#define VP 72   // Vt pitch (halves): [hd][key]; 144 B rows, 16B-aligned

__global__ __launch_bounds__(256) void attn_kernel(
    const float* __restrict__ q, const float* __restrict__ k,
    const float* __restrict__ v, float* __restrict__ o) {
  __shared__ alignas(16) unsigned short Kt[AKT * KP];
  __shared__ alignas(16) unsigned short Vt[HDIM * VP];
  __shared__ alignas(16) unsigned short Ps[8 * 16 * AKT];  // per-wave P scratch
  const int tid = threadIdx.x, lane = tid & 31, wave = tid >> 5;
  const int head = blockIdx.y, kvh = head >> 2;
  const int qbase = blockIdx.x * AQ;
  const int hs = lane >> 4, col0 = lane & 15;
  const float scale = 0.08838834764831845f;  // HD^-0.5

  // Q fragments (pre-scaled), kept in registers for the whole kv loop
  v16bf qa[4];
  {
    const float* qrow = q + (size_t)(qbase + wave * 16 + col0) * HH + head * HDIM;
#pragma unroll
    for (int f = 0; f < 4; ++f) {
      v8u fr;
#pragma unroll
      for (int d = 0; d < 4; ++d) {
        float2 x = *(const float2*)(qrow + f * 32 + 8 * hs + 2 * d);
        fr[d] = pack2bf(x.x * scale, x.y * scale);
        float2 y = *(const float2*)(qrow + f * 32 + 16 + 8 * hs + 2 * d);
        fr[d + 4] = pack2bf(y.x * scale, y.y * scale);
      }
      qa[f] = __builtin_bit_cast(v16bf, fr);
    }
  }

  v8f o_acc[8];
#pragma unroll
  for (int nt = 0; nt < 8; ++nt)
#pragma unroll
    for (int i = 0; i < 8; ++i) o_acc[nt][i] = 0.f;
  float mrow[8], lrow[8];
#pragma unroll
  for (int i = 0; i < 8; ++i) { mrow[i] = -1e30f; lrow[i] = 0.f; }

  const int nkt = qbase / AKT + 2;  // causal: tiles up to qbase+127
  for (int kt = 0; kt < nkt; ++kt) {
    const int kbase = kt * AKT;
    // Stage K (key-major) and V (transposed to hd-major), fp32 -> bf16
#pragma unroll
    for (int it = 0; it < 8; ++it) {
      int slot = tid + it * 256;  // 64 rows x 32 float4
      int row = slot >> 5, c4 = slot & 31;
      const float4 k4 = *(const float4*)(k + (size_t)(kbase + row) * (NKVH * HDIM) + kvh * HDIM + c4 * 4);
      v2u pk = {pack2bf(k4.x, k4.y), pack2bf(k4.z, k4.w)};
      *(v2u*)&Kt[row * KP + c4 * 4] = pk;
      const float4 v4 = *(const float4*)(v + (size_t)(kbase + row) * (NKVH * HDIM) + kvh * HDIM + c4 * 4);
      Vt[(c4 * 4 + 0) * VP + row] = f2bf(v4.x);
      Vt[(c4 * 4 + 1) * VP + row] = f2bf(v4.y);
      Vt[(c4 * 4 + 2) * VP + row] = f2bf(v4.z);
      Vt[(c4 * 4 + 3) * VP + row] = f2bf(v4.w);
    }
    __syncthreads();

    // S = Q @ K^T : 4 key n-tiles x 4 hd k-steps
    v8f s[4];
#pragma unroll
    for (int sn = 0; sn < 4; ++sn) {
#pragma unroll
      for (int i = 0; i < 8; ++i) s[sn][i] = 0.f;
#pragma unroll
      for (int f = 0; f < 4; ++f) {
        v16bf kb = lds_frag_b(Kt, sn * 16, KP, f * 32);
        s[sn] = WMMA_BF16(qa[f], kb, s[sn]);
      }
    }

    // Online softmax (row reductions across the 16 lanes sharing a row)
    float scl[8];
#pragma unroll
    for (int i = 0; i < 8; ++i) {
      int qrow = qbase + wave * 16 + i + 8 * hs;
      float mx = mrow[i];
#pragma unroll
      for (int sn = 0; sn < 4; ++sn) {
        float sv = s[sn][i];
        sv = (kbase + sn * 16 + col0 <= qrow) ? sv : -1e30f;
        s[sn][i] = sv;
        mx = fmaxf(mx, sv);
      }
#pragma unroll
      for (int off = 1; off < 16; off <<= 1) mx = fmaxf(mx, __shfl_xor(mx, off, 32));
      float e = __expf(mrow[i] - mx);
      mrow[i] = mx;
      float rs = 0.f;
#pragma unroll
      for (int sn = 0; sn < 4; ++sn) {
        float pv = __expf(s[sn][i] - mx);
        rs += pv;
        Ps[(wave * 16 + i + 8 * hs) * AKT + sn * 16 + col0] = f2bf(pv);
      }
#pragma unroll
      for (int off = 1; off < 16; off <<= 1) rs += __shfl_xor(rs, off, 32);
      lrow[i] = lrow[i] * e + rs;
      scl[i] = e;
    }
#pragma unroll
    for (int nt = 0; nt < 8; ++nt)
#pragma unroll
      for (int i = 0; i < 8; ++i) o_acc[nt][i] *= scl[i];

    // Wave-private LDS round-trip: wait for our DS writes, then reload as A-fragments
    asm volatile("s_wait_dscnt 0" ::: "memory");
    v16bf pa0 = lds_frag_a(Ps, wave * 16, AKT, 0);
    v16bf pa1 = lds_frag_a(Ps, wave * 16, AKT, 32);
#pragma unroll
    for (int nt = 0; nt < 8; ++nt) {
      v16bf vb0 = lds_frag_b(Vt, nt * 16, VP, 0);
      o_acc[nt] = WMMA_BF16(pa0, vb0, o_acc[nt]);
      v16bf vb1 = lds_frag_b(Vt, nt * 16, VP, 32);
      o_acc[nt] = WMMA_BF16(pa1, vb1, o_acc[nt]);
    }
    __syncthreads();
  }

  float rinv[8];
#pragma unroll
  for (int i = 0; i < 8; ++i) rinv[i] = 1.f / lrow[i];
#pragma unroll
  for (int nt = 0; nt < 8; ++nt)
#pragma unroll
    for (int i = 0; i < 8; ++i) {
      int row = qbase + wave * 16 + i + 8 * hs;
      o[(size_t)row * HH + head * HDIM + nt * 16 + col0] = o_acc[nt][i] * rinv[i];
    }
}

// ---------------------------------------------------------------------------
// Fused (optional add) + RMSNorm. One block per row of H=2048, 256 threads.
// ---------------------------------------------------------------------------
__global__ __launch_bounds__(256) void add_rmsnorm_kernel(
    const float* __restrict__ x, const float* __restrict__ r,
    const float* __restrict__ w, float* __restrict__ res, float* __restrict__ h) {
  int row = blockIdx.x;
  int base = threadIdx.x * 8;
  const float* xr = x + (size_t)row * HH;
  float vals[8];
  float ss = 0.f;
#pragma unroll
  for (int j = 0; j < 8; ++j) {
    float vv = xr[base + j];
    if (r) vv += r[(size_t)row * HH + base + j];
    vals[j] = vv;
    ss += vv * vv;
  }
  if (res) {
#pragma unroll
    for (int j = 0; j < 8; ++j) res[(size_t)row * HH + base + j] = vals[j];
  }
#pragma unroll
  for (int off = 16; off >= 1; off >>= 1) ss += __shfl_xor(ss, off, 32);
  __shared__ float sred[8];
  __shared__ float rsv;
  if ((threadIdx.x & 31) == 0) sred[threadIdx.x >> 5] = ss;
  __syncthreads();
  if (threadIdx.x == 0) {
    float t = 0.f;
    for (int wv = 0; wv < 8; ++wv) t += sred[wv];
    rsv = rsqrtf(t / (float)HH + 1e-5f);
  }
  __syncthreads();
  float rs = rsv;
#pragma unroll
  for (int j = 0; j < 8; ++j)
    h[(size_t)row * HH + base + j] = vals[j] * rs * w[base + j];
}

// NeoX RoPE, in place. One thread per (t, head, j<64).
__global__ __launch_bounds__(256) void rope_kernel(const int* __restrict__ pos,
                                                   float* __restrict__ x, int nheads) {
  int idx = blockIdx.x * 256 + threadIdx.x;
  int j = idx & 63;
  int hh = (idx >> 6) % nheads;
  int t = idx / (64 * nheads);
  // inv_freq = THETA^(-j/64); ln(1e6) = 13.815510557964274
  float inv = __expf(-(float)j * (13.815510557964274f / 64.f));
  float ang = (float)pos[t] * inv;
  float c = cosf(ang), s = sinf(ang);
  float* p = x + (size_t)t * ((size_t)nheads * HDIM) + hh * HDIM + j;
  float x1 = p[0], x2 = p[64];
  p[0] = x1 * c - x2 * s;
  p[64] = x2 * c + x1 * s;
}

// Gate: logits = h2 @ gate_w [2048,8]; softmax fp32; top-2; renormalize; comb[T,8]
__global__ __launch_bounds__(256) void gate_kernel(const float* __restrict__ h2,
                                                   const float* __restrict__ gw,
                                                   float* __restrict__ comb) {
  int t = blockIdx.x;
  float acc[NEXP];
#pragma unroll
  for (int e = 0; e < NEXP; ++e) acc[e] = 0.f;
  for (int j = threadIdx.x; j < HH; j += 256) {
    float x = h2[(size_t)t * HH + j];
    const float* g = gw + (size_t)j * NEXP;
#pragma unroll
    for (int e = 0; e < NEXP; ++e) acc[e] += x * g[e];
  }
#pragma unroll
  for (int e = 0; e < NEXP; ++e)
#pragma unroll
    for (int off = 16; off >= 1; off >>= 1) acc[e] += __shfl_xor(acc[e], off, 32);
  __shared__ float part[8][NEXP];
  if ((threadIdx.x & 31) == 0)
#pragma unroll
    for (int e = 0; e < NEXP; ++e) part[threadIdx.x >> 5][e] = acc[e];
  __syncthreads();
  if (threadIdx.x == 0) {
    float l[NEXP];
    float mx = -1e30f;
#pragma unroll
    for (int e = 0; e < NEXP; ++e) {
      float s = 0.f;
      for (int wv = 0; wv < 8; ++wv) s += part[wv][e];
      l[e] = s;
      mx = fmaxf(mx, s);
    }
    float p[NEXP], sum = 0.f;
#pragma unroll
    for (int e = 0; e < NEXP; ++e) { p[e] = __expf(l[e] - mx); sum += p[e]; }
#pragma unroll
    for (int e = 0; e < NEXP; ++e) p[e] /= sum;
    int i1 = 0; float b1 = p[0];
    for (int e = 1; e < NEXP; ++e) if (p[e] > b1) { b1 = p[e]; i1 = e; }
    int i2 = -1; float b2 = -1.f;
    for (int e = 0; e < NEXP; ++e) if (e != i1 && p[e] > b2) { b2 = p[e]; i2 = e; }
    float den = b1 + b2;
    float outp[NEXP];
#pragma unroll
    for (int e = 0; e < NEXP; ++e) outp[e] = 0.f;
    outp[i1] = b1 / den;
    outp[i2] += b2 / den;
#pragma unroll
    for (int e = 0; e < NEXP; ++e) comb[(size_t)t * NEXP + e] = outp[e];
  }
}

__global__ __launch_bounds__(256) void silu_mul_kernel(float* __restrict__ g1,
                                                       const float* __restrict__ g3, int n) {
  int i = blockIdx.x * 256 + threadIdx.x;
  if (i < n) {
    float x = g1[i];
    float s = x / (1.f + __expf(-x));
    g1[i] = s * g3[i];
  }
}

__global__ __launch_bounds__(256) void zero_kernel(float* __restrict__ p, int n) {
  int i = blockIdx.x * 256 + threadIdx.x;
  if (i < n) p[i] = 0.f;
}

// ---------------------------------------------------------------------------
extern "C" void kernel_launch(void* const* d_in, const int* in_sizes, int n_in,
                              void* d_out, int out_size, void* d_ws, size_t ws_size,
                              hipStream_t stream) {
  const int*   pos      = (const int*)d_in[0];
  const float* hidden   = (const float*)d_in[1];
  const float* residual = (const float*)d_in[2];
  const float* ln1      = (const float*)d_in[3];
  const float* ln2      = (const float*)d_in[4];
  const float* wq       = (const float*)d_in[5];
  const float* wk       = (const float*)d_in[6];
  const float* wv       = (const float*)d_in[7];
  const float* wo       = (const float*)d_in[8];
  const float* gw       = (const float*)d_in[9];
  const float* w1       = (const float*)d_in[10];
  const float* w3       = (const float*)d_in[11];
  const float* w2       = (const float*)d_in[12];

  const size_t TH = (size_t)TT * HH;  // 4M floats
  float* out  = (float*)d_out;        // [T,H]
  float* res2 = out + TH;             // [T,H] second tuple element

  float* ws   = (float*)d_ws;         // ~156 MB of fp32 scratch used
  float* res  = ws;                   // [T,H]
  float* h    = res + TH;             // [T,H]
  float* qb   = h + TH;               // [T, NH*HD]
  float* kb   = qb + TH;              // [T, NKV*HD]
  float* vb   = kb + (size_t)TT * NKVH * HDIM;
  float* ab   = vb + (size_t)TT * NKVH * HDIM;  // attn out [T,H]
  float* h2   = ab + TH;              // [T,H]
  float* comb = h2 + TH;              // [T,8]
  float* g1   = comb + (size_t)TT * NEXP;       // [T,DFF]
  float* g3   = g1 + (size_t)TT * DFF;          // [T,DFF]

  dim3 blk(256);

  // res = hidden + residual ; h = rmsnorm(res) * ln1
  add_rmsnorm_kernel<<<TT, blk, 0, stream>>>(hidden, residual, ln1, res, h);

  // QKV projections (bf16 WMMA)
  gemm_kernel<<<dim3(HH / BN, TT / BM), blk, 0, stream>>>(h, wq, qb, TT, HH, HH, 0, nullptr, nullptr, 0);
  gemm_kernel<<<dim3((NKVH * HDIM) / BN, TT / BM), blk, 0, stream>>>(h, wk, kb, TT, NKVH * HDIM, HH, 0, nullptr, nullptr, 0);
  gemm_kernel<<<dim3((NKVH * HDIM) / BN, TT / BM), blk, 0, stream>>>(h, wv, vb, TT, NKVH * HDIM, HH, 0, nullptr, nullptr, 0);

  // RoPE on q and k
  rope_kernel<<<(TT * NHEADS * 64) / 256, blk, 0, stream>>>(pos, qb, NHEADS);
  rope_kernel<<<(TT * NKVH * 64) / 256, blk, 0, stream>>>(pos, kb, NKVH);

  // Flash attention (causal, GQA)
  attn_kernel<<<dim3(TT / AQ, NHEADS), blk, 0, stream>>>(qb, kb, vb, ab);

  // res2 = attn @ wo + res
  gemm_kernel<<<dim3(HH / BN, TT / BM), blk, 0, stream>>>(ab, wo, res2, TT, HH, HH, 1, res, nullptr, 0);

  // h2 = rmsnorm(res2) * ln2
  add_rmsnorm_kernel<<<TT, blk, 0, stream>>>(res2, nullptr, ln2, nullptr, h2);

  // MoE gate -> per-token combine weights
  gate_kernel<<<TT, blk, 0, stream>>>(h2, gw, comb);

  // out = 0, then accumulate experts
  zero_kernel<<<(int)(TH / 256), blk, 0, stream>>>(out, (int)TH);
  for (int e = 0; e < NEXP; ++e) {
    const float* w1e = w1 + (size_t)e * HH * DFF;
    const float* w3e = w3 + (size_t)e * HH * DFF;
    const float* w2e = w2 + (size_t)e * DFF * HH;
    gemm_kernel<<<dim3(DFF / BN, TT / BM), blk, 0, stream>>>(h2, w1e, g1, TT, DFF, HH, 0, nullptr, nullptr, 0);
    gemm_kernel<<<dim3(DFF / BN, TT / BM), blk, 0, stream>>>(h2, w3e, g3, TT, DFF, HH, 0, nullptr, nullptr, 0);
    silu_mul_kernel<<<(TT * DFF) / 256, blk, 0, stream>>>(g1, g3, TT * DFF);
    gemm_kernel<<<dim3(HH / BN, TT / BM), blk, 0, stream>>>(g1, w2e, out, TT, HH, DFF, 2, nullptr, comb + e, NEXP);
  }
}
